// HierarchicalDiffusion_46600395162218
// MI455X (gfx1250) — matrix operand order repaired and checked by reading, not compiled, not run on hardware
//
#include <hip/hip_runtime.h>

// Problem constants (fixed by the reference setup).
#define NB 8      // batch
#define BN 2048   // nodes
#define BD 128    // embedding dim
#define BK 64     // clusters
#define HSTR 72   // LDS row stride (f16 elements) for transposed h chunk; 72*2=144B, 16B aligned rows

typedef __attribute__((ext_vector_type(16))) _Float16 v16h;
typedef __attribute__((ext_vector_type(8)))  _Float16 v8h;
typedef __attribute__((ext_vector_type(8)))  float    v8f;

// ---------------------------------------------------------------------------
// 0) Zero the global mark buffer (atomicOr target).
// ---------------------------------------------------------------------------
__global__ void zero_ws_kernel(unsigned int* __restrict__ p, int n) {
    int i = blockIdx.x * blockDim.x + threadIdx.x;
    if (i < n) p[i] = 0u;
}

// ---------------------------------------------------------------------------
// 1) Per-cluster node counts via LDS histogram (one block per batch).
// ---------------------------------------------------------------------------
__global__ void counts_kernel(const int* __restrict__ asg, int* __restrict__ counts) {
    __shared__ int cnt[BK];
    int b = blockIdx.x, t = threadIdx.x;
    if (t < BK) cnt[t] = 0;
    __syncthreads();
    for (int j = 0; j < BN / 256; ++j) {
        int a = asg[b * BN + j * 256 + t];
        atomicAdd(&cnt[a], 1);
    }
    __syncthreads();
    if (t < BK) counts[b * BK + t] = cnt[t];
}

// ---------------------------------------------------------------------------
// 2) Streaming scatter-OR over adj (the 134 MB bandwidth-bound part).
//    Block = 16 rows x 2048 cols of one batch; LDS 64x64 byte mark table.
// ---------------------------------------------------------------------------
__global__ void adj_scatter_kernel(const float* __restrict__ adj,
                                   const int* __restrict__ asg,
                                   unsigned int* __restrict__ marks) {
    __shared__ __align__(16) unsigned char asgL[BN];
    __shared__ unsigned char mk[BK * BK];
    int t  = threadIdx.x;
    int b  = blockIdx.x >> 7;    // 128 row-tiles per batch
    int rt = blockIdx.x & 127;
    int r0 = rt * 16;

    for (int j = 0; j < 16; ++j) mk[j * 256 + t] = 0;
    for (int j = 0; j < 8; ++j)  asgL[j * 256 + t] = (unsigned char)asg[b * BN + j * 256 + t];
    __syncthreads();

    const float4* adjB = (const float4*)(adj + (size_t)b * BN * BN);
    for (int p = 0; p < 32; ++p) {
        int idx = p * 256 + t;          // 0..8191 float4s in the 16x2048 tile
        int row = idx >> 9;             // /512 float4s per row
        int c4  = idx & 511;
        size_t off = (size_t)(r0 + row) * 512 + c4;
        __builtin_prefetch(&adjB[off + 2048], 0, 0);   // global_prefetch_b8, ~4 rows ahead
        float4 v = adjB[off];
        int ka = asgL[r0 + row];
        unsigned cb4 = *(const unsigned int*)&asgL[c4 * 4];  // 4 column cluster ids
        unsigned char* mrow = &mk[ka * BK];
        if (v.x > 0.5f) mrow[cb4 & 0xffu]         = 1;
        if (v.y > 0.5f) mrow[(cb4 >> 8) & 0xffu]  = 1;
        if (v.z > 0.5f) mrow[(cb4 >> 16) & 0xffu] = 1;
        if (v.w > 0.5f) mrow[cb4 >> 24]           = 1;
    }
    __syncthreads();

    unsigned int* mb = marks + (size_t)b * BK * BK;
    for (int j = 0; j < 16; ++j) {
        int e = j * 256 + t;
        if (mk[e]) atomicOr(&mb[e], 1u);
    }
}

// ---------------------------------------------------------------------------
// 3) Finalize cluster adjacency: OR with transpose, zero diagonal.
// ---------------------------------------------------------------------------
__global__ void adj_finalize_kernel(const unsigned int* __restrict__ marks,
                                    float* __restrict__ outA) {
    int idx = blockIdx.x * 256 + threadIdx.x;   // < 8*64*64
    int b = idx >> 12;
    int r = idx & 4095;
    int k = r >> 6, l = r & 63;
    unsigned m = marks[(size_t)b * 4096 + k * 64 + l] |
                 marks[(size_t)b * 4096 + l * 64 + k];
    outA[idx] = (k != l && m) ? 1.0f : 0.0f;
}

// ---------------------------------------------------------------------------
// 4) cluster_h via WMMA: sums = onehot^T (K x N) @ h (N x D), then / counts.
//    Block = (batch, 16-cluster tile); wave w of 8 owns d-tile w (16 cols).
//    A (16x32 f16): built in registers from LDS assignments (exact 0/1).
//      Lanes 0-15 hold K=0..15, lanes 16-31 K=16..31 (contiguous, same as B).
//    B (32x16 f16): h chunk staged transposed in LDS so lane L (col n=L%16)
//      reads 16 contiguous f16 at K = (L/16)*16..+15 (contiguous, same as A).
//    C/D (16x16 f32): lane L = col n; VGPR r = row r + 8*(L/16).
// ---------------------------------------------------------------------------
__global__ void cluster_h_wmma_kernel(const float* __restrict__ h,
                                      const int* __restrict__ asg,
                                      const int* __restrict__ counts,
                                      float* __restrict__ outH) {
    __shared__ __align__(16) unsigned short asgL[BN];
    __shared__ __align__(16) _Float16 hT[BD * HSTR];   // transposed chunk: [d][i], i in 0..63

    int t    = threadIdx.x;
    int b    = blockIdx.x >> 2;
    int kt   = blockIdx.x & 3;
    int k0   = kt * 16;
    int lane = t & 31;
    int w    = t >> 5;           // wave id -> d tile
    int d0   = w * 16;
    int m    = lane & 15;        // A row within tile / B,C column
    int half = lane >> 4;

    for (int j = 0; j < BN / 256; ++j)
        asgL[j * 256 + t] = (unsigned short)asg[b * BN + j * 256 + t];

    v8f acc = {};
    const float* hB = h + (size_t)b * BN * BD;
    unsigned tgt = (unsigned)(k0 + m);

    for (int c = 0; c < BN / 64; ++c) {
        int ib = c * 64;
        __syncthreads();   // previous chunk fully consumed (also covers asgL staging)
        {   // stage 64x128 f32 chunk -> LDS f16, transposed (coalesced float4 reads)
            int i  = t >> 5;           // 0..7, + rep*8
            int d4 = (t & 31) * 4;
            for (int rep = 0; rep < 8; ++rep) {
                int ii = rep * 8 + i;
                float4 hv = *(const float4*)(hB + (size_t)(ib + ii) * BD + d4);
                hT[(d4 + 0) * HSTR + ii] = (_Float16)hv.x;
                hT[(d4 + 1) * HSTR + ii] = (_Float16)hv.y;
                hT[(d4 + 2) * HSTR + ii] = (_Float16)hv.z;
                hT[(d4 + 3) * HSTR + ii] = (_Float16)hv.w;
            }
        }
        __syncthreads();
        for (int s = 0; s < 2; ++s) {
            int i0 = ib + s * 32;
            // ---- A tile: compare 16 assignments against this lane's cluster row
            int ab = i0 + half * 16;                      // 16B-aligned (mult of 8 u16)
            uint4 q0 = *(const uint4*)&asgL[ab];          // K = ab..ab+7
            uint4 q1 = *(const uint4*)&asgL[ab + 8];      // K = ab+8..ab+15
            v16h a;
#define PK(d_, w_) \
            a[d_]     = (((w_) & 0xffffu) == tgt) ? (_Float16)1.0f : (_Float16)0.0f; \
            a[(d_)+1] = (((w_) >> 16)     == tgt) ? (_Float16)1.0f : (_Float16)0.0f;
            PK(0,  q0.x) PK(2,  q0.y) PK(4,  q0.z) PK(6,  q0.w)
            PK(8,  q1.x) PK(10, q1.y) PK(12, q1.z) PK(14, q1.w)
#undef PK
            // ---- B tile: 16 contiguous f16 from transposed LDS chunk
            const _Float16* bp = &hT[(d0 + m) * HSTR + s * 32 + half * 16];
            v8h b0 = *(const v8h*)bp;
            v8h b1 = *(const v8h*)(bp + 8);
            v16h bm = __builtin_shufflevector(b0, b1,
                        0, 1, 2, 3, 4, 5, 6, 7, 8, 9, 10, 11, 12, 13, 14, 15);
            acc = __builtin_amdgcn_wmma_f32_16x16x32_f16(
                      false, a, false, bm, (short)0, acc, false, false);
        }
    }

    // Epilogue: divide by count (segment mean; empty clusters stay zero since sum==0)
    float* ob = outH + (size_t)b * BK * BD;
    for (int r = 0; r < 8; ++r) {
        int kr = k0 + r + half * 8;
        int cnt = counts[b * BK + kr];
        float denom = (float)(cnt < 1 ? 1 : cnt);
        ob[(size_t)kr * BD + d0 + m] = acc[r] / denom;
    }
}

// ---------------------------------------------------------------------------
extern "C" void kernel_launch(void* const* d_in, const int* in_sizes, int n_in,
                              void* d_out, int out_size, void* d_ws, size_t ws_size,
                              hipStream_t stream) {
    (void)in_sizes; (void)n_in; (void)out_size; (void)ws_size;
    const float* h   = (const float*)d_in[0];
    const float* adj = (const float*)d_in[1];
    const int*   asg = (const int*)d_in[2];   // assignments (harness integer convention)
    // d_in[3] = num_clusters scalar; fixed at 64 here.

    float* outH = (float*)d_out;                 // (B,K,D) = 65536 floats
    float* outA = outH + (size_t)NB * BK * BD;   // (B,K,K) = 32768 floats

    unsigned int* wsMarks  = (unsigned int*)d_ws;            // B*K*K u32 = 128 KB
    int*          wsCounts = (int*)(wsMarks + NB * BK * BK); // B*K ints

    zero_ws_kernel   <<<(NB * BK * BK + 255) / 256, 256, 0, stream>>>(wsMarks, NB * BK * BK);
    counts_kernel    <<<NB, 256, 0, stream>>>(asg, wsCounts);
    adj_scatter_kernel<<<NB * (BN / 16), 256, 0, stream>>>(adj, asg, wsMarks);
    adj_finalize_kernel<<<(NB * BK * BK) / 256, 256, 0, stream>>>(wsMarks, outA);
    cluster_h_wmma_kernel<<<NB * 4, 256, 0, stream>>>(h, asg, wsCounts, outH);
}
